// Attention_78597901516959
// MI455X (gfx1250) — matrix operand order repaired and checked by reading, not compiled, not run on hardware
//
#include <hip/hip_runtime.h>

typedef __attribute__((ext_vector_type(16))) _Float16 v16h;
typedef __attribute__((ext_vector_type(8)))  _Float16 v8h;
typedef __attribute__((ext_vector_type(4)))  _Float16 v4h;
typedef __attribute__((ext_vector_type(8)))  float    v8f;
typedef __attribute__((ext_vector_type(4)))  float    v4f;

#define SEQ    4096
#define NH     16
#define HD     64
#define QTILE  64      // q rows per block (4 waves x 16)
#define KSTEP  32      // keys per flash step
#define C_L2E  (0.125f * 1.44269504088896f)   // SCALE * log2(e)

// Build a v16h A-operand from two contiguous 8-half LDS runs.
__device__ __forceinline__ v16h make_a(const _Float16* lo_p, const _Float16* hi_p) {
    v8h lo = *(const v8h*)lo_p;
    v8h hi = *(const v8h*)hi_p;
    v16h a;
    #pragma unroll
    for (int i = 0; i < 8; ++i) { a[i] = lo[i]; a[8 + i] = hi[i]; }
    return a;
}

// ---------------------------------------------------------------------------
// Pre-pass: fp32 K,V -> f16, head-major. K -> kh[h][key][d]; V -> vt[h][d][key]
// (transposed through LDS so global reads and writes both stay coalesced).
// ---------------------------------------------------------------------------
__global__ __launch_bounds__(256)
void cvt_kernel(const float* __restrict__ k, const float* __restrict__ v,
                _Float16* __restrict__ kh, _Float16* __restrict__ vt)
{
    __shared__ __align__(16) _Float16 sT[64 * 64];   // [d][key] tile, 8KB
    const int t    = threadIdx.x;
    const int h    = blockIdx.y;
    const int key0 = blockIdx.x * 64;
    #pragma unroll
    for (int i = 0; i < 4; ++i) {
        const int p  = i * 256 + t;       // float4 index in 64x64 tile
        const int kr = p >> 4;            // key row
        const int d4 = (p & 15) * 4;      // d offset
        const size_t gofs = ((size_t)(key0 + kr) * NH + h) * HD + d4;
        v4f kv = *(const v4f*)(k + gofs);
        v4h kp;
        kp[0] = (_Float16)kv[0]; kp[1] = (_Float16)kv[1];
        kp[2] = (_Float16)kv[2]; kp[3] = (_Float16)kv[3];
        *(v4h*)(kh + ((size_t)h * SEQ + key0 + kr) * HD + d4) = kp;
        v4f vv = *(const v4f*)(v + gofs);
        sT[(d4 + 0) * 64 + kr] = (_Float16)vv[0];
        sT[(d4 + 1) * 64 + kr] = (_Float16)vv[1];
        sT[(d4 + 2) * 64 + kr] = (_Float16)vv[2];
        sT[(d4 + 3) * 64 + kr] = (_Float16)vv[3];
    }
    __syncthreads();
    #pragma unroll
    for (int i = 0; i < 2; ++i) {
        const int cc = i * 256 + t;       // 16B chunk, 512 total
        const int d  = cc >> 3;           // 8 chunks per 64-key row
        const int c8 = (cc & 7) * 8;
        v8h x = *(const v8h*)&sT[d * 64 + c8];
        *(v8h*)(vt + ((size_t)h * HD + d) * SEQ + key0 + c8) = x;
    }
}

// ---------------------------------------------------------------------------
// Fast path: async global->LDS staging (ASYNCcnt), double-buffered tiles.
// ---------------------------------------------------------------------------
__global__ __launch_bounds__(128)
void fa_fwd_async(const float* __restrict__ q, const _Float16* __restrict__ kh,
                  const _Float16* __restrict__ vt, float* __restrict__ out)
{
    __shared__ __align__(32) _Float16 sK [2][KSTEP * HD];   // [key][d]  2x4KB
    __shared__ __align__(32) _Float16 sVt[2][HD * KSTEP];   // [d][key]  2x4KB

    const int tid  = threadIdx.x;
    const int wave = tid >> 5;
    const int lane = tid & 31;
    const int lg   = lane >> 4;
    const int ln   = lane & 15;

    const int h   = blockIdx.y;
    const int q0b = blockIdx.x * QTILE;
    const int q0w = q0b + wave * 16;
    const int qi  = q0w + ln;

    const _Float16* kbase = kh + (size_t)h * SEQ * HD;   // + key*HD
    const _Float16* vbase = vt + (size_t)h * HD * SEQ;   // + d*SEQ + key

    // Issue one tile's async copies: 2x16B for K (contiguous 4KB tile) and
    // 2x16B for V^T (64 rows of 64B), per thread. 4 ASYNCcnt ops per thread.
    auto issue_tile = [&](int kk0, int buf) {
        #pragma unroll
        for (int i = 0; i < 2; ++i) {
            const int cc = i * 128 + tid;                      // 16B chunk id
            const _Float16* g = kbase + (size_t)kk0 * HD + cc * 8;
            const unsigned lofs = (unsigned)(size_t)&sK[buf][cc * 8];
            asm volatile("global_load_async_to_lds_b128 %0, %1, off"
                         :: "v"(lofs), "v"(g) : "memory");
        }
        #pragma unroll
        for (int i = 0; i < 2; ++i) {
            const int cc = i * 128 + tid;
            const int d  = cc >> 2;                            // 4 chunks/row
            const int c8 = (cc & 3) * 8;
            const _Float16* g = vbase + (size_t)d * SEQ + kk0 + c8;
            const unsigned lofs = (unsigned)(size_t)&sVt[buf][d * KSTEP + c8];
            asm volatile("global_load_async_to_lds_b128 %0, %1, off"
                         :: "v"(lofs), "v"(g) : "memory");
        }
    };

    // -------- Q^T as two loop-invariant B-operands (d chunks 0-31, 32-63) ---
    v16h qb[2];
    {
        const float* qrow = q + ((size_t)qi * NH + h) * HD;
        #pragma unroll
        for (int dc = 0; dc < 2; ++dc) {
            const int d0 = dc * 32 + lg * 8;
            v4f f0 = *(const v4f*)(qrow + d0);
            v4f f1 = *(const v4f*)(qrow + d0 + 4);
            v4f f2 = *(const v4f*)(qrow + d0 + 16);
            v4f f3 = *(const v4f*)(qrow + d0 + 20);
            v16h b;
            #pragma unroll
            for (int i = 0; i < 4; ++i) {
                b[i]      = (_Float16)f0[i];
                b[4 + i]  = (_Float16)f1[i];
                b[8 + i]  = (_Float16)f2[i];
                b[12 + i] = (_Float16)f3[i];
            }
            qb[dc] = b;
        }
    }

    v8f o0 = {}, o1 = {}, o2 = {}, o3 = {};
    float m = -1e30f, l = 0.0f;

    const int nsteps = (q0b + QTILE) / KSTEP;   // block-uniform

    issue_tile(0, 0);

    for (int s = 0; s < nsteps; ++s) {
        const int k0  = s * KSTEP;
        const int buf = s & 1;

        if (s + 1 < nsteps) {
            issue_tile(k0 + KSTEP, (s + 1) & 1);   // prefetch next tile
            asm volatile("s_wait_asynccnt 0x4" ::: "memory");  // current landed
        } else {
            asm volatile("s_wait_asynccnt 0x0" ::: "memory");
        }
        __syncthreads();   // all waves' async copies for tile s visible

        if (k0 < q0w + 16) {
            const _Float16* bK = sK[buf];
            const _Float16* bV = sVt[buf];

            // S^T = K @ Q^T : 2 key-tiles, chained over d
            v8f st0 = {}, st1 = {};
            #pragma unroll
            for (int dc = 0; dc < 2; ++dc) {
                const int dlo = dc * 32 + lg * 8;
                v16h a0 = make_a(&bK[(0 * 16 + ln) * HD + dlo],
                                 &bK[(0 * 16 + ln) * HD + dlo + 16]);
                v16h a1 = make_a(&bK[(1 * 16 + ln) * HD + dlo],
                                 &bK[(1 * 16 + ln) * HD + dlo + 16]);
                st0 = __builtin_amdgcn_wmma_f32_16x16x32_f16(false, a0, false, qb[dc],
                                                             (short)0, st0, false, false);
                st1 = __builtin_amdgcn_wmma_f32_16x16x32_f16(false, a1, false, qb[dc],
                                                             (short)0, st1, false, false);
            }

            if (k0 + KSTEP - 1 > q0w) {   // causal mask on diagonal steps
                #pragma unroll
                for (int r = 0; r < 8; ++r) {
                    if (k0 + lg * 8 + r      > qi) st0[r] = -1e30f;
                    if (k0 + 16 + lg * 8 + r > qi) st1[r] = -1e30f;
                }
            }

            // online softmax: in-lane trees + one xor-16 combine
            float rm = -1e30f;
            #pragma unroll
            for (int r = 0; r < 8; ++r) rm = fmaxf(rm, fmaxf(st0[r], st1[r]));
            rm = fmaxf(rm, __shfl_xor(rm, 16, 32));
            const float mn    = fmaxf(m, rm);
            const float alpha = exp2f((m - mn) * C_L2E);
            float rs = 0.0f;
            #pragma unroll
            for (int r = 0; r < 8; ++r) {
                const float p0 = exp2f((st0[r] - mn) * C_L2E);
                const float p1 = exp2f((st1[r] - mn) * C_L2E);
                st0[r] = p0; st1[r] = p1;
                rs += p0 + p1;
            }
            rs += __shfl_xor(rs, 16, 32);
            m = mn;
            l = l * alpha + rs;
            #pragma unroll
            for (int r = 0; r < 8; ++r) {
                o0[r] *= alpha; o1[r] *= alpha; o2[r] *= alpha; o3[r] *= alpha;
            }

            // P^T B-operand: swap 8 values with xor-16 partner
            v16h pb;
            #pragma unroll
            for (int r = 0; r < 8; ++r) {
                const float send = lg ? st0[r] : st1[r];
                const float recv = __shfl_xor(send, 16, 32);
                const float own  = lg ? st1[r] : st0[r];
                const float lo   = lg ? recv : own;
                const float hi   = lg ? own  : recv;
                pb[r]     = (_Float16)lo;
                pb[8 + r] = (_Float16)hi;
            }

            // O^T += V^T @ P^T : 4 d-tiles
            const int klo = lg * 16;
            v16h va0 = make_a(&bV[(0 * 16 + ln) * KSTEP + klo],
                              &bV[(0 * 16 + ln) * KSTEP + klo + 8]);
            v16h va1 = make_a(&bV[(1 * 16 + ln) * KSTEP + klo],
                              &bV[(1 * 16 + ln) * KSTEP + klo + 8]);
            v16h va2 = make_a(&bV[(2 * 16 + ln) * KSTEP + klo],
                              &bV[(2 * 16 + ln) * KSTEP + klo + 8]);
            v16h va3 = make_a(&bV[(3 * 16 + ln) * KSTEP + klo],
                              &bV[(3 * 16 + ln) * KSTEP + klo + 8]);
            o0 = __builtin_amdgcn_wmma_f32_16x16x32_f16(false, va0, false, pb, (short)0, o0, false, false);
            o1 = __builtin_amdgcn_wmma_f32_16x16x32_f16(false, va1, false, pb, (short)0, o1, false, false);
            o2 = __builtin_amdgcn_wmma_f32_16x16x32_f16(false, va2, false, pb, (short)0, o2, false, false);
            o3 = __builtin_amdgcn_wmma_f32_16x16x32_f16(false, va3, false, pb, (short)0, o3, false, false);
        }
        __syncthreads();   // tile s consumed; its buffer may be re-filled
    }

    const float inv = 1.0f / l;
    float* orow = out + (size_t)qi * (NH * HD) + h * HD;
    {
        v4f x;
        #pragma unroll
        for (int n = 0; n < 4; ++n) {
            const v8f* on = (n == 0) ? &o0 : (n == 1) ? &o1 : (n == 2) ? &o2 : &o3;
            x[0] = (*on)[0] * inv; x[1] = (*on)[1] * inv;
            x[2] = (*on)[2] * inv; x[3] = (*on)[3] * inv;
            *(v4f*)(orow + n * 16 + lg * 8) = x;
            x[0] = (*on)[4] * inv; x[1] = (*on)[5] * inv;
            x[2] = (*on)[6] * inv; x[3] = (*on)[7] * inv;
            *(v4f*)(orow + n * 16 + lg * 8 + 4) = x;
        }
    }
}

// ---------------------------------------------------------------------------
// Fallback (no workspace): self-contained staging with VGPR round-trip.
// ---------------------------------------------------------------------------
__global__ __launch_bounds__(128)
void fa_fwd_kernel(const float* __restrict__ q, const float* __restrict__ k,
                   const float* __restrict__ v, float* __restrict__ out)
{
    __shared__ __align__(32) _Float16 sK [KSTEP * HD];
    __shared__ __align__(32) _Float16 sVt[HD * KSTEP];

    const int tid  = threadIdx.x;
    const int wave = tid >> 5;
    const int lane = tid & 31;
    const int lg   = lane >> 4;
    const int ln   = lane & 15;

    const int h   = blockIdx.y;
    const int q0b = blockIdx.x * QTILE;
    const int q0w = q0b + wave * 16;
    const int qi  = q0w + ln;

    v16h qb[2];
    {
        const float* qrow = q + ((size_t)qi * NH + h) * HD;
        #pragma unroll
        for (int dc = 0; dc < 2; ++dc) {
            const int d0 = dc * 32 + lg * 8;
            v4f f0 = *(const v4f*)(qrow + d0);
            v4f f1 = *(const v4f*)(qrow + d0 + 4);
            v4f f2 = *(const v4f*)(qrow + d0 + 16);
            v4f f3 = *(const v4f*)(qrow + d0 + 20);
            v16h b;
            #pragma unroll
            for (int i = 0; i < 4; ++i) {
                b[i]      = (_Float16)f0[i];
                b[4 + i]  = (_Float16)f1[i];
                b[8 + i]  = (_Float16)f2[i];
                b[12 + i] = (_Float16)f3[i];
            }
            qb[dc] = b;
        }
    }

    v8f o0 = {}, o1 = {}, o2 = {}, o3 = {};
    float m = -1e30f, l = 0.0f;
    const int nsteps = (q0b + QTILE) / KSTEP;

    for (int s = 0; s < nsteps; ++s) {
        const int k0 = s * KSTEP;
        #pragma unroll
        for (int i = 0; i < 4; ++i) {
            const int p   = i * 128 + tid;
            const int row = p >> 4;
            const int c4  = (p & 15) * 4;
            const size_t gofs = ((size_t)(k0 + row) * NH + h) * HD + c4;
            v4f kv = *(const v4f*)(k + gofs);
            _Float16* dk = &sK[row * HD + c4];
            dk[0] = (_Float16)kv[0]; dk[1] = (_Float16)kv[1];
            dk[2] = (_Float16)kv[2]; dk[3] = (_Float16)kv[3];
            v4f vv = *(const v4f*)(v + gofs);
            sVt[(c4 + 0) * KSTEP + row] = (_Float16)vv[0];
            sVt[(c4 + 1) * KSTEP + row] = (_Float16)vv[1];
            sVt[(c4 + 2) * KSTEP + row] = (_Float16)vv[2];
            sVt[(c4 + 3) * KSTEP + row] = (_Float16)vv[3];
        }
        __syncthreads();

        if (k0 < q0w + 16) {
            v8f st0 = {}, st1 = {};
            #pragma unroll
            for (int dc = 0; dc < 2; ++dc) {
                const int dlo = dc * 32 + lg * 8;
                v16h a0 = make_a(&sK[(0 * 16 + ln) * HD + dlo],
                                 &sK[(0 * 16 + ln) * HD + dlo + 16]);
                v16h a1 = make_a(&sK[(1 * 16 + ln) * HD + dlo],
                                 &sK[(1 * 16 + ln) * HD + dlo + 16]);
                st0 = __builtin_amdgcn_wmma_f32_16x16x32_f16(false, a0, false, qb[dc],
                                                             (short)0, st0, false, false);
                st1 = __builtin_amdgcn_wmma_f32_16x16x32_f16(false, a1, false, qb[dc],
                                                             (short)0, st1, false, false);
            }
            if (k0 + KSTEP - 1 > q0w) {
                #pragma unroll
                for (int r = 0; r < 8; ++r) {
                    if (k0 + lg * 8 + r      > qi) st0[r] = -1e30f;
                    if (k0 + 16 + lg * 8 + r > qi) st1[r] = -1e30f;
                }
            }
            float rm = -1e30f;
            #pragma unroll
            for (int r = 0; r < 8; ++r) rm = fmaxf(rm, fmaxf(st0[r], st1[r]));
            rm = fmaxf(rm, __shfl_xor(rm, 16, 32));
            const float mn    = fmaxf(m, rm);
            const float alpha = exp2f((m - mn) * C_L2E);
            float rs = 0.0f;
            #pragma unroll
            for (int r = 0; r < 8; ++r) {
                const float p0 = exp2f((st0[r] - mn) * C_L2E);
                const float p1 = exp2f((st1[r] - mn) * C_L2E);
                st0[r] = p0; st1[r] = p1;
                rs += p0 + p1;
            }
            rs += __shfl_xor(rs, 16, 32);
            m = mn;
            l = l * alpha + rs;
            #pragma unroll
            for (int r = 0; r < 8; ++r) {
                o0[r] *= alpha; o1[r] *= alpha; o2[r] *= alpha; o3[r] *= alpha;
            }
            v16h pb;
            #pragma unroll
            for (int r = 0; r < 8; ++r) {
                const float send = lg ? st0[r] : st1[r];
                const float recv = __shfl_xor(send, 16, 32);
                const float own  = lg ? st1[r] : st0[r];
                const float lo   = lg ? recv : own;
                const float hi   = lg ? own  : recv;
                pb[r]     = (_Float16)lo;
                pb[8 + r] = (_Float16)hi;
            }
            const int klo = lg * 16;
            v16h va0 = make_a(&sVt[(0 * 16 + ln) * KSTEP + klo],
                              &sVt[(0 * 16 + ln) * KSTEP + klo + 8]);
            v16h va1 = make_a(&sVt[(1 * 16 + ln) * KSTEP + klo],
                              &sVt[(1 * 16 + ln) * KSTEP + klo + 8]);
            v16h va2 = make_a(&sVt[(2 * 16 + ln) * KSTEP + klo],
                              &sVt[(2 * 16 + ln) * KSTEP + klo + 8]);
            v16h va3 = make_a(&sVt[(3 * 16 + ln) * KSTEP + klo],
                              &sVt[(3 * 16 + ln) * KSTEP + klo + 8]);
            o0 = __builtin_amdgcn_wmma_f32_16x16x32_f16(false, va0, false, pb, (short)0, o0, false, false);
            o1 = __builtin_amdgcn_wmma_f32_16x16x32_f16(false, va1, false, pb, (short)0, o1, false, false);
            o2 = __builtin_amdgcn_wmma_f32_16x16x32_f16(false, va2, false, pb, (short)0, o2, false, false);
            o3 = __builtin_amdgcn_wmma_f32_16x16x32_f16(false, va3, false, pb, (short)0, o3, false, false);
        }
        __syncthreads();
    }

    const float inv = 1.0f / l;
    float* orow = out + (size_t)qi * (NH * HD) + h * HD;
    {
        v4f x;
        #pragma unroll
        for (int n = 0; n < 4; ++n) {
            const v8f* on = (n == 0) ? &o0 : (n == 1) ? &o1 : (n == 2) ? &o2 : &o3;
            x[0] = (*on)[0] * inv; x[1] = (*on)[1] * inv;
            x[2] = (*on)[2] * inv; x[3] = (*on)[3] * inv;
            *(v4f*)(orow + n * 16 + lg * 8) = x;
            x[0] = (*on)[4] * inv; x[1] = (*on)[5] * inv;
            x[2] = (*on)[6] * inv; x[3] = (*on)[7] * inv;
            *(v4f*)(orow + n * 16 + lg * 8 + 4) = x;
        }
    }
}

extern "C" void kernel_launch(void* const* d_in, const int* in_sizes, int n_in,
                              void* d_out, int out_size, void* d_ws, size_t ws_size,
                              hipStream_t stream) {
    const float* q = (const float*)d_in[0];
    const float* k = (const float*)d_in[1];
    const float* v = (const float*)d_in[2];
    float* out = (float*)d_out;
    const size_t HALF_BYTES = (size_t)NH * SEQ * HD * sizeof(_Float16);  // 8MB
    if (ws_size >= 2 * HALF_BYTES) {
        _Float16* kh = (_Float16*)d_ws;
        _Float16* vt = (_Float16*)((char*)d_ws + HALF_BYTES);
        cvt_kernel<<<dim3(SEQ / 64, NH, 1), dim3(256, 1, 1), 0, stream>>>(k, v, kh, vt);
        fa_fwd_async<<<dim3(SEQ / QTILE, NH, 1), dim3(128, 1, 1), 0, stream>>>(q, kh, vt, out);
    } else {
        fa_fwd_kernel<<<dim3(SEQ / QTILE, NH, 1), dim3(128, 1, 1), 0, stream>>>(q, k, v, out);
    }
}
